// SelfAttention_661424964120
// MI455X (gfx1250) — matrix operand order repaired and checked by reading, not compiled, not run on hardware
//
#include <hip/hip_runtime.h>
#include <stdint.h>

// ---------------------------------------------------------------------------
// SAGAN self-attention for MI455X (gfx1250, wave32, WMMA bf16 + TDM)
//   f = Wf x, g = Wg x, h = Wh x ; attn = softmax_n(f^T g) ; out = gamma*hP + x
//   Flash attention, fully transposed so softmax is in-lane:
//     S^T = K Q^T   (keys n in lane elements, queries m in lanes)
//     O^T = V^T P^T (channels c in lane elements, queries m in lanes)
// ---------------------------------------------------------------------------

typedef __attribute__((ext_vector_type(16))) __bf16 v16bf;
typedef __attribute__((ext_vector_type(8)))  float  v8f;
typedef __attribute__((ext_vector_type(4)))  unsigned int u32x4;
typedef __attribute__((ext_vector_type(8)))  int i32x8;
typedef __attribute__((ext_vector_type(4)))  int i32x4;

#define N_TOK 4096
#define C_DIM 256
#define D_K   32
#define BM    128          // queries per workgroup (8 waves x 16)
#define BN    32           // keys per block
#define NBLK  (N_TOK / BN)
#define KP    40           // LDS pitch in ushorts (64B row + 16B pad = 80B)

#if __has_builtin(__builtin_amdgcn_tensor_load_to_lds)
#define HAVE_TDM 1
#else
#define HAVE_TDM 0
#endif

// Native f32 -> bf16 conversions (backend emits v_cvt_pk_bf16_f32).
__device__ __forceinline__ unsigned short f32_bf16(float f) {
    union { __bf16 h; unsigned short u; } c;
    c.h = (__bf16)f;
    return c.u;
}
__device__ __forceinline__ unsigned pk_bf16(float lo, float hi) {
    union { __bf16 e[2]; unsigned u; } c;
    c.e[0] = (__bf16)lo; c.e[1] = (__bf16)hi;
    return c.u;
}

union FragBF { v16bf v; unsigned short u[16]; unsigned d[8]; uint4 q[2]; };
union FragF  { v8f v; float f[8]; };

// ---------------------------------------------------------------------------
// TDM: 2D tile load Global->LDS with 16B pad per 64B row (pitch 80B).
// D# bitfields per CDNA5 ISA.
// ---------------------------------------------------------------------------
#if HAVE_TDM
__device__ __forceinline__ void tdm_load_2d(unsigned lds_addr, const void* gptr,
                                            unsigned tensor_d0, unsigned tensor_d1,
                                            unsigned tile_d0, unsigned tile_d1,
                                            unsigned long long stride0)
{
    unsigned long long ga = (unsigned long long)(size_t)gptr;
    u32x4 g0;
    g0[0] = 1u;                                            // count=1, user mode
    g0[1] = lds_addr;                                      // LDS byte address
    g0[2] = (unsigned)(ga & 0xFFFFFFFFu);                  // global_addr[31:0]
    g0[3] = (unsigned)((ga >> 32) & 0x01FFFFFFu) | (2u << 30); // addr[56:32]|type=2
    i32x8 g1;
    // data_size=2B(1), pad_enable, pad_interval=16 DW(64B), pad_amount=4 DW(16B)
    g1[0] = (int)((1u << 16) | (1u << 20) | (3u << 22) | (3u << 25));
    g1[1] = (int)((tensor_d0 & 0xFFFFu) << 16);            // tensor_dim0[15:0]
    g1[2] = (int)((tensor_d0 >> 16) | ((tensor_d1 & 0xFFFFu) << 16));
    g1[3] = (int)((tensor_d1 >> 16) | (tile_d0 << 16));    // | tile_dim0
    g1[4] = (int)(tile_d1 & 0xFFFFu);                      // tile_dim1, tile_dim2=0
    g1[5] = (int)(unsigned)(stride0 & 0xFFFFFFFFull);      // dim0_stride[31:0]
    g1[6] = (int)(unsigned)((stride0 >> 32) & 0xFFFFull);  // dim0_stride[47:32]
    g1[7] = 0;
    i32x4 z = {0, 0, 0, 0};
#if __clang_major__ >= 23
    i32x8 z8 = {0, 0, 0, 0, 0, 0, 0, 0};
    __builtin_amdgcn_tensor_load_to_lds(g0, g1, z, z, z8, 0);
#else
    __builtin_amdgcn_tensor_load_to_lds(g0, g1, z, z, 0);
#endif
}
#endif

// ---------------------------------------------------------------------------
// Elementwise f32 -> bf16 (weights pre-conversion); n must be even.
// ---------------------------------------------------------------------------
__global__ void convert_bf16_kernel(const float* __restrict__ src,
                                    unsigned short* __restrict__ dst, int n) {
    int j = (blockIdx.x * blockDim.x + threadIdx.x) * 2;
    if (j < n) {
        unsigned p = pk_bf16(src[j], src[j + 1]);
        *(unsigned*)(dst + j) = p;
    }
}

// ---------------------------------------------------------------------------
// Projection GEMM, k-outer / row-tile-inner: each workgroup owns a 128-wide
// n-block and computes ALL RT*16 output rows, reusing each x B-fragment RT x.
//   transpose==0: out[b][row][n]          (for h, RT=16)
//   transpose==1: out[b][n][32 rows]      (for fT/gT, RT=2; packed b128 store)
// grid = (N/128, 1, B), block = 256
// ---------------------------------------------------------------------------
template <int RT>
__global__ __launch_bounds__(256, 1) void proj_kernel(
    const unsigned short* __restrict__ Wb,   // bf16 [RT*16][256]
    const float* __restrict__ bias,
    const float* __restrict__ x, unsigned short* __restrict__ out,
    int transpose)
{
    const int b    = blockIdx.z;
    const int lane = threadIdx.x & 31;
    const int wave = threadIdx.x >> 5;
    const int l15  = lane & 15;
    const int hi   = lane >> 4;
    const int n    = blockIdx.x * 128 + wave * 16 + l15;

    const float* xb = x + (size_t)b * C_DIM * N_TOK;
    v8f acc[RT];
    v8f vzero = {};
    #pragma unroll
    for (int rt = 0; rt < RT; ++rt) acc[rt] = vzero;

    for (int k = 0; k < C_DIM; k += 32) {
        // B (x): N = lane&15, K = e (+16 hi half); coalesced across the half-wave
        FragBF bf;
        #pragma unroll
        for (int e = 0; e < 16; e += 2) {
            int kk = k + e + (hi ? 16 : 0);
            float x0 = xb[(size_t)kk * N_TOK + n];
            float x1 = xb[(size_t)(kk + 1) * N_TOK + n];
            bf.d[e >> 1] = pk_bf16(x0, x1);
        }
        #pragma unroll
        for (int rt = 0; rt < RT; ++rt) {
            // A (bf16 weights): two b128 loads in contiguous k = 16*hi + e order, same as B
            FragBF a;
            const unsigned short* wrow = Wb + (size_t)(rt * 16 + l15) * C_DIM + k;
            a.q[0] = *(const uint4*)(wrow + (hi ? 16 : 0));       // K 0..7 / 16..23
            a.q[1] = *(const uint4*)(wrow + 8 + (hi ? 16 : 0));   // K 8..15 / 24..31
            acc[rt] = __builtin_amdgcn_wmma_f32_16x16x32_bf16(false, a.v, false, bf.v,
                                                              (short)0, acc[rt], false, false);
        }
    }

    #pragma unroll
    for (int rt = 0; rt < RT; ++rt) {
        FragF af; af.v = acc[rt];
        const int row0 = rt * 16 + hi * 8;          // C/D layout: M = r + 8*hi
        if (transpose) {                             // out[b][n][row], 32 rows packed
            uint4 st;
            st.x = pk_bf16(af.f[0] + bias[row0 + 0], af.f[1] + bias[row0 + 1]);
            st.y = pk_bf16(af.f[2] + bias[row0 + 2], af.f[3] + bias[row0 + 3]);
            st.z = pk_bf16(af.f[4] + bias[row0 + 4], af.f[5] + bias[row0 + 5]);
            st.w = pk_bf16(af.f[6] + bias[row0 + 6], af.f[7] + bias[row0 + 7]);
            *(uint4*)(out + ((size_t)b * N_TOK + n) * 32 + row0) = st;
        } else {
            #pragma unroll
            for (int r = 0; r < 8; ++r) {
                float val = af.f[r] + bias[row0 + r];
                out[((size_t)b * (RT * 16) + row0 + r) * N_TOK + n] = f32_bf16(val);
            }
        }
    }
}

// ---------------------------------------------------------------------------
// Flash attention (transposed): grid = (N/BM, B), block = 256 (8 waves)
// Each wave owns 16 queries (m = lane&15) and accumulates O^T[256][16].
// Single barrier per key block: wait-tensorcnt -> barrier -> issue-next.
// ---------------------------------------------------------------------------
__global__ __launch_bounds__(256, 1) void attn_kernel(
    const unsigned short* __restrict__ fT,   // keys    [B][N][32]
    const unsigned short* __restrict__ gT,   // queries [B][N][32]
    const unsigned short* __restrict__ hw,   // values  [B][256][N]
    const float* __restrict__ x,
    const float* __restrict__ gamma_p,
    float* __restrict__ out)
{
    __shared__ unsigned short Klds[2][BN * KP];       // K block [n][k], 80B pitch
    __shared__ unsigned short Vlds[2][C_DIM * KP];    // V block [c][n], 80B pitch
    __shared__ unsigned short Plds[8 * 16 * KP];      // per-wave P^T as [m][n]

    const int b    = blockIdx.y;
    const int m0   = blockIdx.x * BM;
    const int tid  = threadIdx.x;
    const int lane = tid & 31;
    const int wave = tid >> 5;
    const int l15  = lane & 15;
    const int hi   = lane >> 4;
    const int mrow = m0 + wave * 16 + l15;

    const unsigned short* fTb = fT + (size_t)b * N_TOK * D_K;
    const unsigned short* hb  = hw + (size_t)b * C_DIM * N_TOK;

    // Q^T as 32x16 bf16 B-fragment: lanes<16 hold k=0..7,16..23, lanes>=16 k=8..15,24..31
    FragBF qb;
    {
        const unsigned short* qp = gT + ((size_t)b * N_TOK + mrow) * D_K + (hi ? 8 : 0);
        qb.q[0] = *(const uint4*)(qp);
        qb.q[1] = *(const uint4*)(qp + 16);
    }

    v8f acc[16];                 // O^T: rows c = ct*16 + r + 8*hi, col m = l15
    v8f vzero = {};
    #pragma unroll
    for (int t = 0; t < 16; ++t) acc[t] = vzero;
    float rmax = -3.0e38f, rsum = 0.0f;   // per-lane scalars (query m = l15)

#if HAVE_TDM
    const unsigned kbase = (unsigned)(unsigned long long)(size_t)(void*)&Klds[0][0];
    const unsigned vbase = (unsigned)(unsigned long long)(size_t)(void*)&Vlds[0][0];
    const unsigned kbufsz = BN * KP * 2u, vbufsz = C_DIM * KP * 2u;
    if (wave == 0) {  // prologue: DMA first K/V block into buffer 0
        tdm_load_2d(kbase, fTb, D_K, BN, D_K, BN, D_K);
        tdm_load_2d(vbase, hb, N_TOK, C_DIM, BN, C_DIM, N_TOK);
    }
#endif

    for (int it = 0; it < NBLK; ++it) {
        const int n0  = it * BN;
        const int buf = HAVE_TDM ? (it & 1) : 0;
#if HAVE_TDM
#if __has_builtin(__builtin_amdgcn_s_wait_tensorcnt)
        if (wave == 0) __builtin_amdgcn_s_wait_tensorcnt((short)0);  // block it landed
#endif
        __syncthreads();   // publishes buf[it&1]; also: all waves done reading buf[(it+1)&1]
        if (wave == 0 && it + 1 < NBLK) {   // DMA next block into the other buffer
            const int nb = (it + 1) & 1;
            tdm_load_2d(kbase + nb * kbufsz,
                        fTb + (size_t)(n0 + BN) * D_K, D_K, BN, D_K, BN, D_K);
            tdm_load_2d(vbase + nb * vbufsz,
                        hb + n0 + BN, N_TOK, C_DIM, BN, C_DIM, N_TOK);
        }
#else
        __syncthreads();
        {   // cooperative staging fallback
            int row = tid >> 3, seg = tid & 7;   // K: 32 rows x 64B
            const uint2* ks = (const uint2*)(fTb + (size_t)(n0 + row) * D_K) + seg;
            *((uint2*)(&Klds[0][row * KP]) + seg) = *ks;
            const uint4* vs = (const uint4*)(hb + (size_t)tid * N_TOK + n0);
            uint4* vd = (uint4*)&Vlds[0][tid * KP];
            vd[0] = vs[0]; vd[1] = vs[1];
        }
        __syncthreads();
#endif
        const unsigned short* kl = &Klds[buf][0];
        const unsigned short* vl = &Vlds[buf][0];

        // ---- S^T = K Q^T : two 16(n)x16(m) subtiles; A = K rows from LDS ----
        FragBF ka0, ka1;
        {
            const unsigned short* p0 = kl + (0 * 16 + l15) * KP;
            const unsigned short* p1 = kl + (1 * 16 + l15) * KP;
            ka0.q[0] = *(const uint4*)(p0 + (hi ? 8 : 0));        // k 0..7 / 8..15
            ka0.q[1] = *(const uint4*)(p0 + 16 + (hi ? 8 : 0));   // k 16..23 / 24..31
            ka1.q[0] = *(const uint4*)(p1 + (hi ? 8 : 0));
            ka1.q[1] = *(const uint4*)(p1 + 16 + (hi ? 8 : 0));
        }
        v8f s0 = __builtin_amdgcn_wmma_f32_16x16x32_bf16(false, ka0.v, false, qb.v,
                                                         (short)0, vzero, false, false);
        v8f s1 = __builtin_amdgcn_wmma_f32_16x16x32_bf16(false, ka1.v, false, qb.v,
                                                         (short)0, vzero, false, false);

        // ---- online softmax over keys: in-lane tree + one half-swap ----
        FragF sf0, sf1; sf0.v = s0; sf1.v = s1;
        float mx = sf0.f[0];
        #pragma unroll
        for (int r = 1; r < 8; ++r) mx = fmaxf(mx, sf0.f[r]);
        #pragma unroll
        for (int r = 0; r < 8; ++r) mx = fmaxf(mx, sf1.f[r]);
        mx = fmaxf(mx, __shfl_xor(mx, 16, 32));     // combine the two lane halves
        const float mnew  = fmaxf(rmax, mx);
        const float alpha = __expf(rmax - mnew);
        rmax = mnew;

        float pv0[8], pv1[8];
        #pragma unroll
        for (int r = 0; r < 8; ++r) pv0[r] = __expf(sf0.f[r] - mnew);
        #pragma unroll
        for (int r = 0; r < 8; ++r) pv1[r] = __expf(sf1.f[r] - mnew);
        float ps = 0.0f;
        #pragma unroll
        for (int r = 0; r < 8; ++r) ps += pv0[r] + pv1[r];
        ps += __shfl_xor(ps, 16, 32);
        rsum = rsum * alpha + ps;

        if (__any(alpha < 1.0f)) {   // rescale only when some lane's max moved
            #pragma unroll
            for (int t = 0; t < 16; ++t) {
                FragF a; a.v = acc[t];
                #pragma unroll
                for (int r = 0; r < 8; ++r) a.f[r] *= alpha;
                acc[t] = a.v;
            }
        }

        // ---- P^T -> bf16 in LDS [m][n]: one b128 store per subtile ----
        unsigned short* pw = &Plds[wave * 16 * KP] + l15 * KP;
        {
            uint4 st0, st1;
            st0.x = pk_bf16(pv0[0], pv0[1]); st0.y = pk_bf16(pv0[2], pv0[3]);
            st0.z = pk_bf16(pv0[4], pv0[5]); st0.w = pk_bf16(pv0[6], pv0[7]);
            st1.x = pk_bf16(pv1[0], pv1[1]); st1.y = pk_bf16(pv1[2], pv1[3]);
            st1.z = pk_bf16(pv1[4], pv1[5]); st1.w = pk_bf16(pv1[6], pv1[7]);
            *(uint4*)(pw + hi * 8)      = st0;   // subtile 0: n = 8*hi .. 8*hi+7
            *(uint4*)(pw + 16 + hi * 8) = st1;   // subtile 1: n = 16 + 8*hi ..
        }
        FragBF pb;   // P^T B-fragment (lanes<16: n=0..7,16..23, lanes>=16: n=8..15,24..31)
        {            // DS ops in-order within a wave: no barrier needed
            pb.q[0] = *(const uint4*)(pw + (hi ? 8 : 0));
            pb.q[1] = *(const uint4*)(pw + 16 + (hi ? 8 : 0));
        }

        // ---- O^T += V^T P^T : 16 c-tiles; A = V rows from LDS ----
        #pragma unroll
        for (int ct = 0; ct < 16; ++ct) {
            FragBF va;
            const unsigned short* vp = vl + (ct * 16 + l15) * KP;
            va.q[0] = *(const uint4*)(vp + (hi ? 8 : 0));        // n 0..7 / 8..15
            va.q[1] = *(const uint4*)(vp + 16 + (hi ? 8 : 0));   // n 16..23 / 24..31
            acc[ct] = __builtin_amdgcn_wmma_f32_16x16x32_bf16(false, va.v, false, pb.v,
                                                              (short)0, acc[ct], false, false);
        }
    }

    // ---- epilogue: out[c][m] = gamma*(acc/l) + x ; coalesced across lanes ----
    const float scale = gamma_p[0] / rsum;
    const float* xb = x   + (size_t)b * C_DIM * N_TOK;
    float*       ob = out + (size_t)b * C_DIM * N_TOK;
    #pragma unroll
    for (int ct = 0; ct < 16; ++ct) {
        FragF a; a.v = acc[ct];
        #pragma unroll
        for (int r = 0; r < 8; ++r) {
            const int c = ct * 16 + r + hi * 8;
            const size_t idx = (size_t)c * N_TOK + mrow;
            ob[idx] = fmaf(scale, a.f[r], xb[idx]);
        }
    }
}

// ---------------------------------------------------------------------------
extern "C" void kernel_launch(void* const* d_in, const int* in_sizes, int n_in,
                              void* d_out, int out_size, void* d_ws, size_t ws_size,
                              hipStream_t stream) {
    const float* x     = (const float*)d_in[0];
    const float* Wf    = (const float*)d_in[1];
    const float* bfp   = (const float*)d_in[2];
    const float* Wg    = (const float*)d_in[3];
    const float* bgp   = (const float*)d_in[4];
    const float* Wh    = (const float*)d_in[5];
    const float* bhp   = (const float*)d_in[6];
    const float* gamma = (const float*)d_in[7];
    float* out = (float*)d_out;

    unsigned short* fTw = (unsigned short*)d_ws;                 // [4][4096][32] bf16
    unsigned short* gTw = fTw + (size_t)4 * N_TOK * D_K;         // [4][4096][32] bf16
    unsigned short* hws = gTw + (size_t)4 * N_TOK * D_K;         // [4][256][4096] bf16
    unsigned short* wfb = hws + (size_t)4 * C_DIM * N_TOK;       // [32][256] bf16
    unsigned short* wgb = wfb + (size_t)D_K * C_DIM;             // [32][256] bf16
    unsigned short* whb = wgb + (size_t)D_K * C_DIM;             // [256][256] bf16

    dim3 blk(256);
    convert_bf16_kernel<<<dim3(16),  blk, 0, stream>>>(Wf, wfb, D_K * C_DIM);
    convert_bf16_kernel<<<dim3(16),  blk, 0, stream>>>(Wg, wgb, D_K * C_DIM);
    convert_bf16_kernel<<<dim3(128), blk, 0, stream>>>(Wh, whb, C_DIM * C_DIM);

    proj_kernel<2> <<<dim3(N_TOK / 128, 1, 4), blk, 0, stream>>>(wfb, bfp, x, fTw, 1);
    proj_kernel<2> <<<dim3(N_TOK / 128, 1, 4), blk, 0, stream>>>(wgb, bgp, x, gTw, 1);
    proj_kernel<16><<<dim3(N_TOK / 128, 1, 4), blk, 0, stream>>>(whb, bhp, x, hws, 0);
    attn_kernel<<<dim3(N_TOK / BM, 4), blk, 0, stream>>>(fTw, gTw, hws, x, gamma, out);
}
